// StatefulCausalHopfieldAttention_12558484374041
// MI455X (gfx1250) — matrix operand order repaired and checked by reading, not compiled, not run on hardware
//
#include <hip/hip_runtime.h>

// ---------------- problem constants ----------------
#define H_     16
#define DIN_   1024
#define DH_    64
#define SEQ_   1024
#define STATE_ 128
#define T_     (SEQ_ + 2 * STATE_)   // 1280
#define P_     T_                    // 1280
#define SCALE_ 0.125f                // 1/sqrt(64)

typedef __bf16 bf16;
typedef __attribute__((ext_vector_type(16))) __bf16 v16bf;
typedef __attribute__((ext_vector_type(8)))  float  v8f;

// ---------------- WMMA helpers (CDNA5 16x16x32 bf16) ----------------
static __device__ __forceinline__ v8f wmma_bf16(v16bf a, v16bf b, v8f c) {
  return __builtin_amdgcn_wmma_f32_16x16x32_bf16(
      /*neg_a=*/false, a, /*neg_b=*/false, b,
      /*c_mod=*/(short)0, c, /*reuse_a=*/false, /*reuse_b=*/false);
}

// A matrix 16(M) x 32(K), row-major source, ld in elements.
// Contiguous k order used here (B matches): lanes 0-15 hold M=lane, K={0..15}; lanes 16-31 K={16..31}.
static __device__ __forceinline__ v16bf load_A(const bf16* base, int ld, int k0) {
  int lane = threadIdx.x & 31;
  int m = lane & 15, hi = lane >> 4;
  const bf16* row = base + (size_t)m * ld + k0 + hi * 16;
  v16bf a;
#pragma unroll
  for (int e = 0; e < 16; ++e) a[e] = row[e];
  return a;
}

// B matrix 32(K) x 16(N) where B[k][n] = src[n][k]  (src row-major N x K).
// 32B contiguous per lane -> 2x global_load_b128.
static __device__ __forceinline__ v16bf load_Bt(const bf16* src, int ld, int n0, int k0) {
  int lane = threadIdx.x & 31;
  int n = lane & 15, hi = lane >> 4;
  const bf16* p = src + (size_t)(n0 + n) * ld + k0 + hi * 16;
  v16bf b;
#pragma unroll
  for (int e = 0; e < 16; ++e) b[e] = p[e];
  return b;
}

// C/D 16x16 f32: VGPR r -> row (hi*8 + r), lane&15 -> col
static __device__ __forceinline__ void store_C_bf(bf16* dst, int ld, v8f c, int n0) {
  int lane = threadIdx.x & 31;
  int n = lane & 15, mhi = (lane >> 4) * 8;
#pragma unroll
  for (int r = 0; r < 8; ++r) dst[(size_t)(mhi + r) * ld + n0 + n] = (bf16)c[r];
}

// transposed store: dst[(n0+n)][t0+row]
static __device__ __forceinline__ void store_C_bf_T(bf16* dst, int ld, v8f c, int n0, int t0) {
  int lane = threadIdx.x & 31;
  int n = lane & 15, mhi = (lane >> 4) * 8;
#pragma unroll
  for (int r = 0; r < 8; ++r) dst[(size_t)(n0 + n) * ld + t0 + mhi + r] = (bf16)c[r];
}

// ---------------- conversion kernels ----------------
__global__ void cvt_kernel(const float* __restrict__ in, bf16* __restrict__ out, int n) {
  int i = blockIdx.x * blockDim.x + threadIdx.x;
  if (i < n) out[i] = (bf16)in[i];
}

// batched transpose + convert: in[b][r][c] (f32) -> out[b][c][r] (bf16)
__global__ void cvtT_kernel(const float* __restrict__ in, bf16* __restrict__ out,
                            int R, int C, int total) {
  int i = blockIdx.x * blockDim.x + threadIdx.x;
  if (i < total) {
    int b = i / (R * C);
    int rc = i - b * (R * C);
    int r = rc / C, c = rc - r * C;
    out[(size_t)b * R * C + (size_t)c * R + r] = (bf16)in[i];
  }
}

// ---------------- GEMM kernels ----------------
// mem[h][t][:] = x[t][:] @ W[h]; W stored transposed wT[h][n][k], ld=DIN.
// Software-pipelined: next K-step's fragments are loaded before this step's WMMAs.
__global__ void __launch_bounds__(32)
hidden_proj_kernel(const bf16* __restrict__ xb, const bf16* __restrict__ whT,
                   const bf16* __restrict__ whsT, bf16* __restrict__ memb) {
  int h = blockIdx.x;
  int t0 = blockIdx.y * 16;
  const bf16* w = ((t0 < STATE_) || (t0 >= T_ - STATE_)) ? whsT : whT;
  w += (size_t)h * DIN_ * DH_;
  const bf16* xbase = xb + (size_t)t0 * DIN_;

  v16bf a  = load_A(xbase, DIN_, 0);
  v16bf b0 = load_Bt(w, DIN_, 0,  0);
  v16bf b1 = load_Bt(w, DIN_, 16, 0);
  v16bf b2 = load_Bt(w, DIN_, 32, 0);
  v16bf b3 = load_Bt(w, DIN_, 48, 0);
  v8f c0 = {}, c1 = {}, c2 = {}, c3 = {};
  for (int k0 = 0; k0 < DIN_; k0 += 32) {
    int kn = (k0 + 32 < DIN_) ? (k0 + 32) : k0;
    v16bf an  = load_A(xbase, DIN_, kn);
    v16bf nb0 = load_Bt(w, DIN_, 0,  kn);
    v16bf nb1 = load_Bt(w, DIN_, 16, kn);
    v16bf nb2 = load_Bt(w, DIN_, 32, kn);
    v16bf nb3 = load_Bt(w, DIN_, 48, kn);
    c0 = wmma_bf16(a, b0, c0);
    c1 = wmma_bf16(a, b1, c1);
    c2 = wmma_bf16(a, b2, c2);
    c3 = wmma_bf16(a, b3, c3);
    a = an; b0 = nb0; b1 = nb1; b2 = nb2; b3 = nb3;
  }
  bf16* out = memb + ((size_t)h * T_ + t0) * DH_;
  store_C_bf(out, DH_, c0, 0);
  store_C_bf(out, DH_, c1, 16);
  store_C_bf(out, DH_, c2, 32);
  store_C_bf(out, DH_, c3, 48);
}

// out[h][t][:] = in[h][t][:] @ W[h] (wT[h][n][k], ld=DH). All 10 fragments
// preloaded, then 8 back-to-back WMMAs.
__global__ void __launch_bounds__(32)
proj64_kernel(const bf16* __restrict__ inb, const bf16* __restrict__ wstateT,
              const bf16* __restrict__ wmidT, bf16* __restrict__ outb,
              bf16* __restrict__ outbT) {
  int h = blockIdx.x;
  int t0 = blockIdx.y * 16;
  const bf16* w = ((t0 < STATE_) || (t0 >= T_ - STATE_)) ? wstateT : wmidT;
  w += (size_t)h * DH_ * DH_;
  const bf16* A = inb + ((size_t)h * T_ + t0) * DH_;

  v16bf a0  = load_A(A, DH_, 0),  a1  = load_A(A, DH_, 32);
  v16bf b00 = load_Bt(w, DH_, 0, 0),  b01 = load_Bt(w, DH_, 16, 0);
  v16bf b02 = load_Bt(w, DH_, 32, 0), b03 = load_Bt(w, DH_, 48, 0);
  v16bf b10 = load_Bt(w, DH_, 0, 32),  b11 = load_Bt(w, DH_, 16, 32);
  v16bf b12 = load_Bt(w, DH_, 32, 32), b13 = load_Bt(w, DH_, 48, 32);

  v8f c0 = {}, c1 = {}, c2 = {}, c3 = {};
  c0 = wmma_bf16(a0, b00, c0); c1 = wmma_bf16(a0, b01, c1);
  c2 = wmma_bf16(a0, b02, c2); c3 = wmma_bf16(a0, b03, c3);
  c0 = wmma_bf16(a1, b10, c0); c1 = wmma_bf16(a1, b11, c1);
  c2 = wmma_bf16(a1, b12, c2); c3 = wmma_bf16(a1, b13, c3);

  if (outb) {
    bf16* out = outb + ((size_t)h * T_ + t0) * DH_;
    store_C_bf(out, DH_, c0, 0);
    store_C_bf(out, DH_, c1, 16);
    store_C_bf(out, DH_, c2, 32);
    store_C_bf(out, DH_, c3, 48);
  }
  if (outbT) {
    bf16* outT = outbT + (size_t)h * DH_ * T_;   // [d][t], ld = T_
    store_C_bf_T(outT, T_, c0, 0,  t0);
    store_C_bf_T(outT, T_, c1, 16, t0);
    store_C_bf_T(outT, T_, c2, 32, t0);
    store_C_bf_T(outT, T_, c3, 48, t0);
  }
}

// ---------------- fused CoPE-gated causal attention ----------------
// One wave per (head, 16-row q tile). LDS: 16x1280 bf16 CoPE-logit gather
// table + 16x32 bf16 P-tile staging. V consumed d-major, loaded at the start of each pair; K/CoPE loads
// software-pipelined one tile ahead of their WMMAs.
__global__ void __launch_bounds__(32)
attn_kernel(const bf16* __restrict__ qb, const bf16* __restrict__ kb,
            const bf16* __restrict__ vtb, const bf16* __restrict__ copeTb,
            bf16* __restrict__ out_bf, float* __restrict__ out_f32) {
  __shared__ bf16 sLint[16 * P_];   // 40 KB
  __shared__ bf16 sPt[16 * 32];     // 1 KB

  int h  = blockIdx.x;
  int q0 = blockIdx.y * 16;
  int lane = threadIdx.x & 31;
  int n = lane & 15, hi = lane >> 4, mhi = hi * 8;

  const bf16* Q = qb + ((size_t)h * T_ + q0) * DH_;
  v16bf aq0 = load_A(Q, DH_, 0);
  v16bf aq1 = load_A(Q, DH_, 32);

  // ---- CoPE-logit tile: Lint[m][p] = q_m . cope[:,p]; copeT is [p][d] ----
  {
    v16bf cb0 = load_Bt(copeTb, DH_, 0, 0);
    v16bf cb1 = load_Bt(copeTb, DH_, 0, 32);
    for (int pt = 0; pt < P_ / 16; ++pt) {
      int ptn = (pt + 1 < P_ / 16) ? (pt + 1) : pt;
      v16bf nb0 = load_Bt(copeTb, DH_, ptn * 16, 0);
      v16bf nb1 = load_Bt(copeTb, DH_, ptn * 16, 32);
      v8f c = {};
      c = wmma_bf16(aq0, cb0, c);
      c = wmma_bf16(aq1, cb1, c);
#pragma unroll
      for (int r = 0; r < 8; ++r) sLint[(mhi + r) * P_ + pt * 16 + n] = (bf16)c[r];
      cb0 = nb0; cb1 = nb1;
    }
  }
  __syncthreads();

  const bf16* K  = kb  + (size_t)h * T_ * DH_;   // [t][d]
  const bf16* Vt = vtb + (size_t)h * DH_ * T_;   // [d][t]

  // ---- pass A: row totals of sigmoid gates over all k ----
  float tot[8];
#pragma unroll
  for (int r = 0; r < 8; ++r) tot[r] = 0.f;
  {
    v16bf kb0 = load_Bt(K, DH_, 0, 0);
    v16bf kb1 = load_Bt(K, DH_, 0, 32);
    for (int kt = 0; kt < T_ / 16; ++kt) {
      int ktn = (kt + 1 < T_ / 16) ? (kt + 1) : kt;
      v16bf nb0 = load_Bt(K, DH_, ktn * 16, 0);
      v16bf nb1 = load_Bt(K, DH_, ktn * 16, 32);
      v8f c = {};
      c = wmma_bf16(aq0, kb0, c);
      c = wmma_bf16(aq1, kb1, c);
#pragma unroll
      for (int r = 0; r < 8; ++r) tot[r] += 1.f / (1.f + __expf(-c[r]));
      kb0 = nb0; kb1 = nb1;
    }
  }
#pragma unroll
  for (int r = 0; r < 8; ++r)
    for (int s = 1; s < 16; s <<= 1) tot[r] += __shfl_xor(tot[r], s, 32);

  // ---- pass B: scores + online softmax + P.V via WMMA ----
  float pref[8], mrow[8], lrow[8];
#pragma unroll
  for (int r = 0; r < 8; ++r) { pref[r] = 0.f; mrow[r] = -3.0e38f; lrow[r] = 0.f; }
  v8f o0 = {}, o1 = {}, o2 = {}, o3 = {};

  // K fragments for the first tile pair
  v16bf kf00 = load_Bt(K, DH_, 0, 0),  kf01 = load_Bt(K, DH_, 0, 32);
  v16bf kf10 = load_Bt(K, DH_, 16, 0), kf11 = load_Bt(K, DH_, 16, 32);

  for (int kp = 0; kp < T_ / 32; ++kp) {
    // V fragments for this pair: issued now, consumed after the softmax VALU.
    v16bf vb0 = load_Bt(Vt, T_, 0,  kp * 32);
    v16bf vb1 = load_Bt(Vt, T_, 16, kp * 32);
    v16bf vb2 = load_Bt(Vt, T_, 32, kp * 32);
    v16bf vb3 = load_Bt(Vt, T_, 48, kp * 32);

    v8f sc[2];
#pragma unroll
    for (int sub = 0; sub < 2; ++sub) {
      int kt = kp * 2 + sub;
      v8f c = {};
      if (sub == 0) {
        c = wmma_bf16(aq0, kf00, c);
        c = wmma_bf16(aq1, kf01, c);
      } else {
        c = wmma_bf16(aq0, kf10, c);
        c = wmma_bf16(aq1, kf11, c);
      }
      // refill this sub's K fragments for the next pair; latency hidden
      // under the scan/softmax VALU below.
      int ktn = (kt + 2 < T_ / 16) ? (kt + 2) : kt;
      if (sub == 0) {
        kf00 = load_Bt(K, DH_, ktn * 16, 0);
        kf01 = load_Bt(K, DH_, ktn * 16, 32);
      } else {
        kf10 = load_Bt(K, DH_, ktn * 16, 0);
        kf11 = load_Bt(K, DH_, ktn * 16, 32);
      }
#pragma unroll
      for (int r = 0; r < 8; ++r) {
        float logit = c[r];
        float g = 1.f / (1.f + __expf(-logit));
        // inclusive scan across 16-lane segment (k-column dimension)
        float incl = g;
        for (int s = 1; s < 16; s <<= 1) {
          float u = __shfl_up(incl, (unsigned)s, 16);
          if ((lane & 15) >= s) incl += u;
        }
        float excl = incl - g;
        float tsum = __shfl(incl, 15, 16);
        float pos = tot[r] - (pref[r] + excl);
        pref[r] += tsum;
        pos = fminf(pos, (float)(P_ - 1));
        pos = fmaxf(pos, 0.f);
        float pf = floorf(pos);
        int ipf = (int)pf;
        int ipc = (int)ceilf(pos);
        float wfrac = pos - pf;
        int m = mhi + r;
        float lf = (float)sLint[m * P_ + ipf];
        float lc = (float)sLint[m * P_ + ipc];
        float bias = lc * wfrac + lf * (1.f - wfrac);
        float score = logit * SCALE_ + bias;
        int kg = kt * 16 + n;
        if (kg > q0 + m) score = -3.0e38f;   // causal mask
        c[r] = score;
      }
      sc[sub] = c;
    }
    // online softmax over the 32-wide pair
#pragma unroll
    for (int r = 0; r < 8; ++r) {
      float rmax = fmaxf(sc[0][r], sc[1][r]);
      for (int s = 8; s >= 1; s >>= 1) rmax = fmaxf(rmax, __shfl_xor(rmax, s, 32));
      float mnew = fmaxf(mrow[r], rmax);
      float a = __expf(mrow[r] - mnew);
      float p0 = __expf(sc[0][r] - mnew);
      float p1 = __expf(sc[1][r] - mnew);
      float psum = p0 + p1;
      for (int s = 8; s >= 1; s >>= 1) psum += __shfl_xor(psum, s, 32);
      lrow[r] = lrow[r] * a + psum;
      mrow[r] = mnew;
      o0[r] *= a; o1[r] *= a; o2[r] *= a; o3[r] *= a;
      int m = mhi + r;
      sPt[m * 32 + n]      = (bf16)p0;
      sPt[m * 32 + 16 + n] = (bf16)p1;
    }
    __syncthreads();
    v16bf ap = load_A(sPt, 32, 0);
    o0 = wmma_bf16(ap, vb0, o0);
    o1 = wmma_bf16(ap, vb1, o1);
    o2 = wmma_bf16(ap, vb2, o2);
    o3 = wmma_bf16(ap, vb3, o3);
    __syncthreads();
  }

  // ---- epilogue: normalize and store ----
#pragma unroll
  for (int r = 0; r < 8; ++r) {
    int m = mhi + r;
    float inv = 1.f / lrow[r];
    size_t base = ((size_t)h * T_ + q0 + m) * DH_;
    if (out_f32) {
      float* od = out_f32 + base;
      od[n] = o0[r] * inv; od[16 + n] = o1[r] * inv;
      od[32 + n] = o2[r] * inv; od[48 + n] = o3[r] * inv;
    }
    if (out_bf) {
      bf16* od = out_bf + base;
      od[n] = (bf16)(o0[r] * inv); od[16 + n] = (bf16)(o1[r] * inv);
      od[32 + n] = (bf16)(o2[r] * inv); od[48 + n] = (bf16)(o3[r] * inv);
    }
  }
}

// ---------------- host launch ----------------
extern "C" void kernel_launch(void* const* d_in, const int* in_sizes, int n_in,
                              void* d_out, int out_size, void* d_ws, size_t ws_size,
                              hipStream_t stream) {
  (void)in_sizes; (void)n_in; (void)out_size; (void)ws_size;
  const float* x    = (const float*)d_in[0];
  const float* Wh   = (const float*)d_in[1];
  const float* Whs  = (const float*)d_in[2];
  const float* Wq   = (const float*)d_in[3];
  const float* Wk   = (const float*)d_in[4];
  const float* Wv   = (const float*)d_in[5];
  const float* Wqs  = (const float*)d_in[6];
  const float* Wks  = (const float*)d_in[7];
  const float* Wvs  = (const float*)d_in[8];
  const float* cope = (const float*)d_in[9];
  float* out = (float*)d_out;

  char* wsp = (char*)d_ws;
  auto alloc = [&](size_t elems) {
    bf16* p = (bf16*)wsp;
    wsp += ((elems * sizeof(bf16) + 255) / 256) * 256;
    return p;
  };
  const size_t NX = (size_t)T_ * DIN_;
  const size_t NWH = (size_t)H_ * DIN_ * DH_;
  const size_t NWS = (size_t)H_ * DH_ * DH_;
  const size_t NCOPE = (size_t)DH_ * P_;
  const size_t NM = (size_t)H_ * T_ * DH_;

  bf16* xb     = alloc(NX);
  bf16* whT    = alloc(NWH);     // [h][n][k] k-fast
  bf16* whsT   = alloc(NWH);
  bf16* wqT    = alloc(NWS);
  bf16* wkT    = alloc(NWS);
  bf16* wvT    = alloc(NWS);
  bf16* wqsT   = alloc(NWS);
  bf16* wksT   = alloc(NWS);
  bf16* wvsT   = alloc(NWS);
  bf16* copeT  = alloc(NCOPE);   // [p][d]
  bf16* memb   = alloc(NM);
  bf16* kbuf   = alloc(NM);      // [h][t][d]
  bf16* vbufT  = alloc(NM);      // [h][d][t]
  bf16* qbuf   = alloc(NM);
  bf16* mem1b  = alloc(NM);

  auto cv = [&](const float* src, bf16* dst, size_t nEl) {
    cvt_kernel<<<dim3((unsigned)((nEl + 255) / 256)), dim3(256), 0, stream>>>(src, dst, (int)nEl);
  };
  auto cvT = [&](const float* src, bf16* dst, int R, int C, size_t nEl) {
    cvtT_kernel<<<dim3((unsigned)((nEl + 255) / 256)), dim3(256), 0, stream>>>(src, dst, R, C, (int)nEl);
  };
  cv(x, xb, NX);
  cvT(Wh,  whT,  DIN_, DH_, NWH);
  cvT(Whs, whsT, DIN_, DH_, NWH);
  cvT(Wq,  wqT,  DH_, DH_, NWS);
  cvT(Wk,  wkT,  DH_, DH_, NWS);
  cvT(Wv,  wvT,  DH_, DH_, NWS);
  cvT(Wqs, wqsT, DH_, DH_, NWS);
  cvT(Wks, wksT, DH_, DH_, NWS);
  cvT(Wvs, wvsT, DH_, DH_, NWS);
  cvT(cope, copeT, DH_, P_, NCOPE);

  dim3 gTiles(H_, T_ / 16), b32(32);
  // mem0 = proj(x, W_hidden / W_hidden_state)
  hidden_proj_kernel<<<gTiles, b32, 0, stream>>>(xb, whT, whsT, memb);
  // k (row-major), v (d-major transposed), q for iter 0
  proj64_kernel<<<gTiles, b32, 0, stream>>>(memb, wksT, wkT, kbuf,  nullptr);
  proj64_kernel<<<gTiles, b32, 0, stream>>>(memb, wvsT, wvT, nullptr, vbufT);
  proj64_kernel<<<gTiles, b32, 0, stream>>>(memb, wqsT, wqT, qbuf,  nullptr);
  // iter 0 -> mem1 (bf16, feeds next q projection)
  attn_kernel<<<gTiles, b32, 0, stream>>>(qbuf, kbuf, vbufT, copeT, mem1b, nullptr);
  // iter 1 q from mem1
  proj64_kernel<<<gTiles, b32, 0, stream>>>(mem1b, wqsT, wqT, qbuf, nullptr);
  // iter 1 -> final f32 output
  attn_kernel<<<gTiles, b32, 0, stream>>>(qbuf, kbuf, vbufT, copeT, nullptr, out);
}
